// SSFM_3564822855868
// MI455X (gfx1250) — compile-verified, matrix operand order checked by reading, not run on hardware
//
#include <hip/hip_runtime.h>
#include <hip/hip_bf16.h>

// SSFM fusion block for gfx1250 (MI455X).
// Memory-bound (~41 GFLOP vs ~280 MB traffic @ 23.3 TB/s), so:
//  - all large intermediates are f16 (halves their HBM traffic),
//  - all GEMM-shaped ops use v_wmma_f32_16x16x32_f16 (f32 accumulate),
//  - k-contiguous f16 tiles stage via global_load_async_to_lds_b128
//    (ASYNCcnt / s_wait_asynccnt), fp32 tiles via float4 + packed v4h stores,
//  - bias / att+proj adds / BN / ReLU fused into GEMM epilogues,
//  - 3x3 fused conv is implicit GEMM with a once-repacked f16 weight tensor.

typedef __attribute__((ext_vector_type(16))) _Float16 v16h;
typedef __attribute__((ext_vector_type(8)))  _Float16 v8h;
typedef __attribute__((ext_vector_type(4)))  _Float16 v4h;
typedef __attribute__((ext_vector_type(8)))  float    v8f;

#define BN_EPS 1e-5f

union V16U { v16h v; v8h h[2]; };
__device__ __forceinline__ v16h mk16(v8h lo, v8h hi) {
    V16U u; u.h[0] = lo; u.h[1] = hi; return u.v;
}

// ---------------------------------------------------------------------------
// Tile staging helpers: 128 rows x 32 k (k contiguous in source) -> LDS f16.
// fp32 source: float4 loads + packed v4h LDS stores (returns false).
// f16 source: async copy straight to LDS, ASYNCcnt-tracked (returns true;
//             caller must s_wait_asynccnt before the barrier).
// ---------------------------------------------------------------------------
__device__ __forceinline__ bool stage_tile(_Float16* lds, int ldsStride,
                                           const float* src, long long ldk, int tid)
{
    #pragma unroll
    for (int j = 0; j < 4; ++j) {
        int ch = tid + j * 256;               // 1024 chunks of 4
        int r = ch >> 3, kc = (ch & 7) * 4;
        float4 v = *(const float4*)(src + (long long)r * ldk + kc);
        v4h h; h[0] = (_Float16)v.x; h[1] = (_Float16)v.y;
               h[2] = (_Float16)v.z; h[3] = (_Float16)v.w;
        *(v4h*)&lds[r * ldsStride + kc] = h;
    }
    return false;
}

__device__ __forceinline__ bool stage_tile(_Float16* lds, int ldsStride,
                                           const _Float16* src, long long ldk, int tid)
{
    #pragma unroll
    for (int j = 0; j < 2; ++j) {
        int ch = tid + j * 256;               // 512 chunks of 8 halfs (16B)
        int r = ch >> 2, kc = (ch & 3) * 8;
        unsigned dst = (unsigned)(size_t)&lds[r * ldsStride + kc];  // LDS offset
        const _Float16* g = src + (long long)r * ldk + kc;
        asm volatile("global_load_async_to_lds_b128 %0, %1, off"
                     :: "v"(dst), "v"(g) : "memory");
    }
    return true;
}

// NN B tile: B row-major [K][N]; write k-transposed lBT[n][k].
__device__ __forceinline__ void stage_tileT(_Float16* lds, const float* B,
                                            int N, int nBase, int tid)
{
    #pragma unroll
    for (int j = 0; j < 4; ++j) {
        int ch = tid + j * 256;
        int k = ch >> 5, nc = (ch & 31) * 4;
        float4 v = *(const float4*)(B + (long long)k * N + nBase + nc);
        lds[(nc + 0) * 40 + k] = (_Float16)v.x;
        lds[(nc + 1) * 40 + k] = (_Float16)v.y;
        lds[(nc + 2) * 40 + k] = (_Float16)v.z;
        lds[(nc + 3) * 40 + k] = (_Float16)v.w;
    }
}
__device__ __forceinline__ void stage_tileT(_Float16* lds, const _Float16* B,
                                            int N, int nBase, int tid)
{
    #pragma unroll
    for (int j = 0; j < 2; ++j) {
        int ch = tid + j * 256;
        int k = ch >> 4, nc = (ch & 15) * 8;
        v8h v = *(const v8h*)(B + (long long)k * N + nBase + nc);
        #pragma unroll
        for (int i = 0; i < 8; ++i) lds[(nc + i) * 40 + k] = v[i];
    }
}

// ---------------------------------------------------------------------------
// Tiled WMMA GEMM:  C[b] = A[b](MxK) * B[b](KxN, or NxK if BT) [+bias][+add]
// 128x128 tile, K-step 32, 256 threads = 8 waves; wave w owns rows
// [16w,16w+16) x 128 cols (8 accumulator tiles). M%128==0, N%128==0, K%32==0.
// ---------------------------------------------------------------------------
template <bool BT, typename TA, typename TB, typename TC>
__global__ __launch_bounds__(256)
void gemm_wmma(const TA* __restrict__ A, long long sAb,
               const TB* __restrict__ B, long long sBb,
               TC* __restrict__ C, long long sCb,
               const float* __restrict__ bias,
               const TC* __restrict__ add, long long sAddb,
               int N, int K)
{
    __shared__ _Float16 lA[128 * 40];
    __shared__ _Float16 lBT[128 * 40];

    const int tid  = threadIdx.x;
    const int wave = tid >> 5, lane = tid & 31;
    const int half = lane >> 4, l15 = lane & 15;
    const int nBase = blockIdx.x * 128;
    const int mBase = blockIdx.y * 128;
    const int b     = blockIdx.z;

    A += (long long)b * sAb;
    B += (long long)b * sBb;
    C += (long long)b * sCb;
    if (add) add += (long long)b * sAddb;

    v8f acc[8] = {};

    for (int k0 = 0; k0 < K; k0 += 32) {
        bool asyncUsed = stage_tile(lA, 40, A + (long long)mBase * K + k0, K, tid);
        if (BT)
            asyncUsed |= stage_tile(lBT, 40, B + (long long)nBase * K + k0, K, tid);
        else
            stage_tileT(lBT, B + (long long)k0 * N, N, nBase, tid);

        if (k0 + 32 < K) {  // prefetch next K-slab (global_prefetch_b8)
            __builtin_prefetch(&A[(long long)(mBase + (tid >> 1)) * K + k0 + 32 + (tid & 1) * 16], 0, 1);
            if (BT)
                __builtin_prefetch(&B[(long long)(nBase + (tid >> 1)) * K + k0 + 32 + (tid & 1) * 16], 0, 1);
            else
                __builtin_prefetch(&B[(long long)(k0 + 32 + (tid >> 3)) * N + nBase + (tid & 7) * 16], 0, 1);
        }
        if (asyncUsed) asm volatile("s_wait_asynccnt 0x0" ::: "memory");
        __syncthreads();

        const int arow = wave * 16 + l15;
        v8h alo = *(const v8h*)&lA[arow * 40 + 8 * half];
        v8h ahi = *(const v8h*)&lA[arow * 40 + 16 + 8 * half];
        v16h afrag = mk16(alo, ahi);
        #pragma unroll
        for (int t = 0; t < 8; ++t) {
            const int col = t * 16 + l15;
            v8h blo = *(const v8h*)&lBT[col * 40 + 8 * half];
            v8h bhi = *(const v8h*)&lBT[col * 40 + 16 + 8 * half];
            acc[t] = __builtin_amdgcn_wmma_f32_16x16x32_f16(
                false, afrag, false, mk16(blo, bhi), (short)0, acc[t], false, false);
        }
        __syncthreads();
    }

    #pragma unroll
    for (int t = 0; t < 8; ++t) {
        const int col = nBase + t * 16 + l15;
        #pragma unroll
        for (int r = 0; r < 8; ++r) {
            const int row = mBase + wave * 16 + r + 8 * half;
            float v = acc[t][r];
            if (bias) v += bias[row];
            long long off = (long long)row * N + col;
            if (add) v += (float)add[off];
            C[off] = (TC)v;
        }
    }
}

// ---------------------------------------------------------------------------
// Depthwise 3x3 conv + bias on f16 [B,C,64,64], pad=1 (plain VALU, mem-bound)
// ---------------------------------------------------------------------------
__global__ void dwconv3x3(const _Float16* __restrict__ in, const float* __restrict__ w,
                          const float* __restrict__ bias, _Float16* __restrict__ out,
                          int C, int total)
{
    int idx = blockIdx.x * 256 + threadIdx.x;
    if (idx >= total) return;
    int x = idx & 63, y = (idx >> 6) & 63;
    int c = (idx >> 12) % C;
    long long base = (long long)(idx >> 12) << 12;
    float s = bias[c];
    #pragma unroll
    for (int dy = 0; dy < 3; ++dy) {
        int yy = y + dy - 1;
        if ((unsigned)yy >= 64u) continue;
        #pragma unroll
        for (int dx = 0; dx < 3; ++dx) {
            int xx = x + dx - 1;
            if ((unsigned)xx >= 64u) continue;
            s += w[c * 9 + dy * 3 + dx] * (float)in[base + yy * 64 + xx];
        }
    }
    out[idx] = (_Float16)s;
}

// ---------------------------------------------------------------------------
// Gate network kernels on fp32 [8, C, 128, 128] (tiny)
// ---------------------------------------------------------------------------
__global__ void gate_conv(const float* __restrict__ in, const float* __restrict__ w,
                          const float* __restrict__ bias,
                          const float* __restrict__ bng, const float* __restrict__ bnb,
                          float* __restrict__ out, int IC, int OC, int sigmoid_out)
{
    int idx = blockIdx.x * 256 + threadIdx.x;
    int total = 8 * OC * 128 * 128;
    if (idx >= total) return;
    int x = idx & 127, y = (idx >> 7) & 127;
    int oc = (idx >> 14) % OC, b = idx / (OC << 14);
    float s = bias[oc];
    for (int ic = 0; ic < IC; ++ic) {
        const float* ip = in + ((long long)(b * IC + ic) << 14);
        const float* wp = w + (oc * IC + ic) * 9;
        #pragma unroll
        for (int dy = 0; dy < 3; ++dy) {
            int yy = y + dy - 1;
            if ((unsigned)yy >= 128u) continue;
            #pragma unroll
            for (int dx = 0; dx < 3; ++dx) {
                int xx = x + dx - 1;
                if ((unsigned)xx >= 128u) continue;
                s += wp[dy * 3 + dx] * ip[yy * 128 + xx];
            }
        }
    }
    float v = s * (bng[oc] * rsqrtf(1.f + BN_EPS)) + bnb[oc];
    out[idx] = sigmoid_out ? (1.f / (1.f + expf(-v))) : v;
}

__global__ void gate_frelu(const float* __restrict__ g, const float* __restrict__ w,
                           const float* __restrict__ bias,
                           const float* __restrict__ bng, const float* __restrict__ bnb,
                           float* __restrict__ out)
{
    int idx = blockIdx.x * 256 + threadIdx.x;
    if (idx >= 8 * 16 * 128 * 128) return;
    int x = idx & 127, y = (idx >> 7) & 127;
    int c = (idx >> 14) & 15;
    const float* ip = g + ((long long)(idx >> 14) << 14);
    float s = bias[c];
    #pragma unroll
    for (int dy = 0; dy < 3; ++dy) {
        int yy = y + dy - 1;
        if ((unsigned)yy >= 128u) continue;
        #pragma unroll
        for (int dx = 0; dx < 3; ++dx) {
            int xx = x + dx - 1;
            if ((unsigned)xx >= 128u) continue;
            s += w[c * 9 + dy * 3 + dx] * ip[yy * 128 + xx];
        }
    }
    float x1 = s * (bng[c] * rsqrtf(1.f + BN_EPS)) + bnb[c];
    out[idx] = fmaxf(g[idx], x1);
}

__global__ void softmax128(const float* __restrict__ in, float* __restrict__ out)
{
    __shared__ float red[128];
    int r = blockIdx.x, t = threadIdx.x;
    float v = in[r * 128 + t];
    red[t] = v; __syncthreads();
    for (int s = 64; s > 0; s >>= 1) {
        if (t < s) red[t] = fmaxf(red[t], red[t + s]);
        __syncthreads();
    }
    float m = red[0]; __syncthreads();
    float e = expf(v - m);
    red[t] = e; __syncthreads();
    for (int s = 64; s > 0; s >>= 1) {
        if (t < s) red[t] += red[t + s];
        __syncthreads();
    }
    out[r * 128 + t] = e / red[0];
}

// Repack fus_w [128][256][3][3] f32 -> [tap][oc][ic] f16 (once, 590 KB)
__global__ void pack_fusw(const float* __restrict__ W, _Float16* __restrict__ out)
{
    int idx = blockIdx.x * 256 + threadIdx.x;
    if (idx >= 9 * 128 * 256) return;
    int tap = idx >> 15;
    int rem = idx & 32767;
    int m = rem >> 8, ic = rem & 255;
    out[idx] = (_Float16)W[(m * 256 + ic) * 9 + tap];
}

// ---------------------------------------------------------------------------
// Fused 3x3 conv (256->128, pad=1) + BN + ReLU as implicit GEMM with WMMA.
// One workgroup per (h, b): M=128 out-channels x N=64 pixels. Per 32-channel
// slab: haloed rows h-1..h+1 staged k-transposed via v8h loads; per-tap
// weight tile async-copied from the repacked f16 weights. 4 wmma/tap/wave.
// ---------------------------------------------------------------------------
__global__ __launch_bounds__(256)
void fusconv3x3_wmma(const _Float16* __restrict__ U, const _Float16* __restrict__ Wp,
                     const float* __restrict__ bng, const float* __restrict__ bnb,
                     float* __restrict__ out)
{
    __shared__ _Float16 lUT[3 * 66 * 32];   // [dy][x(+halo)][k]
    __shared__ _Float16 lW[128 * 32];       // [m][k]

    const int h = blockIdx.x, b = blockIdx.y;
    const int tid  = threadIdx.x;
    const int wave = tid >> 5, lane = tid & 31;
    const int half = lane >> 4, l15 = lane & 15;

    v8f acc[4] = {};
    const _Float16* Ub = U + ((long long)b << 20);   // b * 256 * 4096

    for (int ic0 = 0; ic0 < 256; ic0 += 32) {
        __syncthreads();  // previous iteration's lUT readers done
        for (int e = tid; e < 3 * 32 * 8; e += 256) {    // 8-half chunks
            int xc = (e & 7) * 8, c = (e >> 3) & 31, dy = e >> 8;
            int hh = h + dy - 1;
            if ((unsigned)hh < 64u) {
                v8h v = *(const v8h*)(Ub + (((long long)(ic0 + c)) << 12) + hh * 64 + xc);
                #pragma unroll
                for (int i = 0; i < 8; ++i)
                    lUT[(dy * 66 + xc + i + 1) * 32 + c] = v[i];
            } else {
                #pragma unroll
                for (int i = 0; i < 8; ++i)
                    lUT[(dy * 66 + xc + i + 1) * 32 + c] = (_Float16)0.f;
            }
        }
        for (int e = tid; e < 3 * 2 * 32; e += 256) {    // zero halo columns
            int c = e & 31, wh = (e >> 5) & 1, dy = e >> 6;
            lUT[(dy * 66 + (wh ? 65 : 0)) * 32 + c] = (_Float16)0.f;
        }

        for (int dy = 0; dy < 3; ++dy) {
            for (int dx = 0; dx < 3; ++dx) {
                __syncthreads();  // lW readers from previous tap done
                stage_tile(lW, 32, Wp + (((long long)(dy * 3 + dx)) << 15) + ic0,
                           256, tid);                       // async f16 copy
                asm volatile("s_wait_asynccnt 0x0" ::: "memory");
                __syncthreads();

                int arow = wave * 16 + l15;
                v8h alo = *(const v8h*)&lW[arow * 32 + 8 * half];
                v8h ahi = *(const v8h*)&lW[arow * 32 + 16 + 8 * half];
                v16h afrag = mk16(alo, ahi);
                #pragma unroll
                for (int t = 0; t < 4; ++t) {
                    int xcol = t * 16 + l15 + dx;
                    v8h blo = *(const v8h*)&lUT[(dy * 66 + xcol) * 32 + 8 * half];
                    v8h bhi = *(const v8h*)&lUT[(dy * 66 + xcol) * 32 + 16 + 8 * half];
                    acc[t] = __builtin_amdgcn_wmma_f32_16x16x32_f16(
                        false, afrag, false, mk16(blo, bhi), (short)0, acc[t], false, false);
                }
            }
        }
    }

    const float rs = rsqrtf(1.f + BN_EPS);
    #pragma unroll
    for (int t = 0; t < 4; ++t) {
        int x = t * 16 + l15;
        #pragma unroll
        for (int r = 0; r < 8; ++r) {
            int m = wave * 16 + r + 8 * half;
            float v = acc[t][r] * (bng[m] * rs) + bnb[m];
            out[(((long long)(b * 128 + m)) << 12) + h * 64 + x] = fmaxf(v, 0.f);
        }
    }
}

// ---------------------------------------------------------------------------
extern "C" void kernel_launch(void* const* d_in, const int* in_sizes, int n_in,
                              void* d_out, int out_size, void* d_ws, size_t ws_size,
                              hipStream_t stream)
{
    (void)in_sizes; (void)n_in; (void)out_size; (void)ws_size;
    typedef _Float16 h16;

    const float* rgb          = (const float*)d_in[0];
    const float* depth        = (const float*)d_in[1];
    const float* rgb_pconv_w  = (const float*)d_in[2];
    const float* rgb_pconv_b  = (const float*)d_in[3];
    const float* rgb_dconv_w  = (const float*)d_in[4];
    const float* rgb_dconv_b  = (const float*)d_in[5];
    const float* dep_pconv_w  = (const float*)d_in[6];
    const float* dep_pconv_b  = (const float*)d_in[7];
    const float* dep_dconv_w  = (const float*)d_in[8];
    const float* dep_dconv_b  = (const float*)d_in[9];
    const float* conv1_w      = (const float*)d_in[10];
    const float* conv1_b      = (const float*)d_in[11];
    const float* conv2_w      = (const float*)d_in[12];
    const float* conv2_b      = (const float*)d_in[13];
    const float* g1_w         = (const float*)d_in[14];
    const float* g1_b         = (const float*)d_in[15];
    const float* g1_bn_g      = (const float*)d_in[16];
    const float* g1_bn_b      = (const float*)d_in[17];
    const float* fr_w         = (const float*)d_in[18];
    const float* fr_b         = (const float*)d_in[19];
    const float* fr_bn_g      = (const float*)d_in[20];
    const float* fr_bn_b      = (const float*)d_in[21];
    const float* g2_w         = (const float*)d_in[22];
    const float* g2_b         = (const float*)d_in[23];
    const float* g2_bn_g      = (const float*)d_in[24];
    const float* g2_bn_b      = (const float*)d_in[25];
    const float* fus_w        = (const float*)d_in[26];
    const float* fus_bn_g     = (const float*)d_in[27];
    const float* fus_bn_b     = (const float*)d_in[28];
    float* outp = (float*)d_out;

    // workspace (byte carve-out, 256B-aligned chunks; ~113 MB total)
    const long long HW  = 4096;
    char* wsb = (char*)d_ws;
    auto carve = [&](size_t bytes) {
        char* p = wsb; wsb += (bytes + 255) & ~(size_t)255; return p;
    };
    h16*   tmpP  = (h16*)  carve(8ULL * 384 * HW * 2);   // pconv out, reused as U
    h16*   rt    = (h16*)  carve(8ULL * 384 * HW * 2);
    h16*   dt    = (h16*)  carve(8ULL * 384 * HW * 2);
    float* qk    = (float*)carve(8ULL * 2 * 16384 * 4);
    float* gate1 = (float*)carve(8ULL * 16 * 16384 * 4);
    float* gate2 = (float*)carve(8ULL * 16 * 16384 * 4);
    float* gsig  = (float*)carve(8ULL * 2 * 16384 * 4);
    float* attw  = (float*)carve(8ULL * 2 * 16384 * 4);
    h16*   rgb_p = (h16*)  carve(8ULL * 128 * HW * 2);
    h16*   dep_p = (h16*)  carve(8ULL * 128 * HW * 2);
    h16*   wpack = (h16*)  carve(9ULL * 128 * 256 * 2);
    h16*   U     = tmpP;                                 // [8,256,64,64] f16

    dim3 blk(256);

    // weight repack for the fused 3x3 conv (tiny, done up front)
    pack_fusw<<<1152, blk, 0, stream>>>(fus_w, wpack);

    // 1) pconv rgb: [384,256] x [256,4096] per batch -> f16
    gemm_wmma<false, float, float, h16><<<dim3(32, 3, 8), blk, 0, stream>>>(
        rgb_pconv_w, 0, rgb, 256 * HW, tmpP, 384 * HW,
        rgb_pconv_b, nullptr, 0, 4096, 256);
    // 2) depthwise rgb
    dwconv3x3<<<49152, blk, 0, stream>>>(tmpP, rgb_dconv_w, rgb_dconv_b, rt,
                                         384, 8 * 384 * 4096);
    // 3) pconv depth
    gemm_wmma<false, float, float, h16><<<dim3(32, 3, 8), blk, 0, stream>>>(
        dep_pconv_w, 0, depth, 256 * HW, tmpP, 384 * HW,
        dep_pconv_b, nullptr, 0, 4096, 256);
    // 4) depthwise depth
    dwconv3x3<<<49152, blk, 0, stream>>>(tmpP, dep_dconv_w, dep_dconv_b, dt,
                                         384, 8 * 384 * 4096);

    // 5) rgb_qk = rq @ dk^T  (f16 NT GEMM, async-to-LDS staging)
    gemm_wmma<true, h16, h16, float><<<dim3(1, 1, 8), blk, 0, stream>>>(
        rt, 384 * HW, dt + 128 * HW, 384 * HW, qk, 2LL * 16384,
        nullptr, nullptr, 0, 128, 4096);
    // 6) depth_qk = dq @ rk^T
    gemm_wmma<true, h16, h16, float><<<dim3(1, 1, 8), blk, 0, stream>>>(
        dt, 384 * HW, rt + 128 * HW, 384 * HW, qk + 16384, 2LL * 16384,
        nullptr, nullptr, 0, 128, 4096);

    // 7-9) gate network (fp32, tiny)
    gate_conv<<<8192, blk, 0, stream>>>(qk, g1_w, g1_b, g1_bn_g, g1_bn_b,
                                        gate1, 2, 16, 0);
    gate_frelu<<<8192, blk, 0, stream>>>(gate1, fr_w, fr_b, fr_bn_g, fr_bn_b,
                                         gate2);
    gate_conv<<<1024, blk, 0, stream>>>(gate2, g2_w, g2_b, g2_bn_g, g2_bn_b,
                                        gsig, 16, 2, 1);
    // 10) row softmax
    softmax128<<<2048, 128, 0, stream>>>(gsig, attw);

    // 11-12) 1x1 projections -> f16
    gemm_wmma<false, float, float, h16><<<dim3(32, 1, 8), blk, 0, stream>>>(
        conv1_w, 0, rgb, 256 * HW, rgb_p, 128 * HW,
        conv1_b, nullptr, 0, 4096, 256);
    gemm_wmma<false, float, float, h16><<<dim3(32, 1, 8), blk, 0, stream>>>(
        conv2_w, 0, depth, 256 * HW, dep_p, 128 * HW,
        conv2_b, nullptr, 0, 4096, 256);

    // 13) U[:, :128] = depth_a @ dv + rgb_p   (f32 A, f16 B, f16 C)
    gemm_wmma<false, float, h16, h16><<<dim3(32, 1, 8), blk, 0, stream>>>(
        attw + 16384, 2LL * 16384, dt + 256 * HW, 384 * HW,
        U, 256 * HW, nullptr, rgb_p, 128 * HW, 4096, 128);
    // 14) U[:, 128:] = rgb_a @ rv + dep_p
    gemm_wmma<false, float, h16, h16><<<dim3(32, 1, 8), blk, 0, stream>>>(
        attw, 2LL * 16384, rt + 256 * HW, 384 * HW,
        U + 128 * HW, 256 * HW, nullptr, dep_p, 128 * HW, 4096, 128);

    // 15) fused 3x3 conv + BN + ReLU -> fp32 output
    fusconv3x3_wmma<<<dim3(64, 8), blk, 0, stream>>>(U, wpack, fus_bn_g,
                                                     fus_bn_b, outp);
}
